// SparseVoxelDownsampleModule2_22222160789585
// MI455X (gfx1250) — matrix operand order repaired and checked by reading, not compiled, not run on hardware
//
#include <hip/hip_runtime.h>
#include <hip/hip_bf16.h>

typedef __attribute__((ext_vector_type(16))) __bf16 v16bf;
typedef __attribute__((ext_vector_type(8)))  float  v8f;
typedef __attribute__((ext_vector_type(4)))  int    v4i;

#define EPS_BN 1e-5f

#define AS_GLOBAL __attribute__((address_space(1)))
#define AS_LDS    __attribute__((address_space(3)))

// gfx1250 async load-to-LDS path (probe via __has_builtin; fallback keeps compile green)
#if defined(__HIP_DEVICE_COMPILE__) && __has_builtin(__builtin_amdgcn_global_load_async_to_lds_b128) && __has_builtin(__builtin_amdgcn_s_wait_asynccnt)
#define ASYNC_LDS_OK 1
#else
#define ASYNC_LDS_OK 0
#endif

__device__ __forceinline__ void async_copy16(const void* gsrc, void* lds_dst) {
#if ASYNC_LDS_OK
  __builtin_amdgcn_global_load_async_to_lds_b128((AS_GLOBAL v4i*)gsrc,
                                                 (AS_LDS v4i*)lds_dst, 0, 0);
#else
  (void)gsrc; (void)lds_dst;
#endif
}
__device__ __forceinline__ void wait_async() {
#if ASYNC_LDS_OK
  __builtin_amdgcn_s_wait_asynccnt(0);
#endif
}

// ---------------- utility kernels ----------------

__global__ void k_zero_f32(float* __restrict__ p, long n) {
  long i = (long)blockIdx.x * blockDim.x + threadIdx.x;
  long stride = (long)gridDim.x * blockDim.x;
  for (; i < n; i += stride) p[i] = 0.0f;
}

__global__ void k_batch_sums(const int* __restrict__ coords, float* __restrict__ sums, int n) {
  int i = blockIdx.x * blockDim.x + threadIdx.x;
  if (i >= n) return;
  int b = coords[i * 4 + 0];
  atomicAdd(&sums[b * 4 + 0], (float)coords[i * 4 + 1]);
  atomicAdd(&sums[b * 4 + 1], (float)coords[i * 4 + 2]);
  atomicAdd(&sums[b * 4 + 2], (float)coords[i * 4 + 3]);
  atomicAdd(&sums[b * 4 + 3], 1.0f);
}

__global__ void k_calc_shift(const float* __restrict__ sums, float* __restrict__ shift, int B, int S) {
  int t = threadIdx.x;
  if (t >= B * 3) return;
  int b = t / 3, d = t % 3;
  float cnt = fmaxf(sums[b * 4 + 3], 1.0f);
  shift[t] = (float)(S / 2) - sums[b * 4 + d] / cnt;
}

__global__ void k_scatter(const int* __restrict__ coords, const float* __restrict__ feat,
                          const float* __restrict__ shift, float* __restrict__ dense,
                          float* __restrict__ mask, int n, int S) {
  int i = blockIdx.x * blockDim.x + threadIdx.x;
  if (i >= n) return;
  int b = coords[i * 4 + 0];
  int q[3];
#pragma unroll
  for (int d = 0; d < 3; ++d) {
    float v = (float)coords[i * 4 + 1 + d] + shift[b * 3 + d];
    int qi = (int)rintf(v);  // round-to-nearest-even, matches jnp.round
    qi = qi < 0 ? 0 : (qi > S - 1 ? S - 1 : qi);
    q[d] = qi;
  }
  long idx = (((long)b * S + q[0]) * S + q[1]) * S + q[2];
  atomicAdd(&dense[idx], feat[i]);
  mask[idx] = 1.0f;
}

// mask3 = 3x3x3 SAME max-dilate of mask; also 2x2x2 stride-2 pool mask3 into maskp
__global__ void k_mask3_pool(const float* __restrict__ mask, float* __restrict__ mask3,
                             float* __restrict__ maskp, int S, long nvox) {
  long i = (long)blockIdx.x * blockDim.x + threadIdx.x;
  if (i >= nvox) return;
  int x = (int)(i % S);
  long r = i / S;
  int y = (int)(r % S); r /= S;
  int z = (int)(r % S);
  int b = (int)(r / S);
  float m = 0.0f;
  for (int dz = -1; dz <= 1; ++dz)
    for (int dy = -1; dy <= 1; ++dy)
      for (int dx = -1; dx <= 1; ++dx) {
        int zz = z + dz, yy = y + dy, xx = x + dx;
        if ((unsigned)zz < (unsigned)S && (unsigned)yy < (unsigned)S && (unsigned)xx < (unsigned)S)
          m = fmaxf(m, mask[(((long)b * S + zz) * S + yy) * S + xx]);
      }
  mask3[i] = m;
  int Sh = S >> 1;
  long pidx = (((long)b * Sh + (z >> 1)) * Sh + (y >> 1)) * Sh + (x >> 1);
  atomicMax((unsigned int*)&maskp[pidx], __float_as_uint(m));  // m >= 0
}

// Pack W (3,3,3,cin,cout) f32 into per-lane B-fragment order:
// Wp[((nt*NCHUNK + kc)*32 + lane)*16 + j] = W[k*COUT+n], k = kc*32 + (lane>>4)*8 + j + (j>=8 ? 8 : 0),
// n = nt*16 + (lane&15); zero padded. One chunk = contiguous 1KB.
__global__ void k_pack_w(const float* __restrict__ W, __bf16* __restrict__ Wp,
                         int K, int NCHUNK, int COUT, int NT) {
  long i = (long)blockIdx.x * blockDim.x + threadIdx.x;
  long tot = (long)NT * NCHUNK * 512;
  if (i >= tot) return;
  int j = (int)(i & 15);
  int lane = (int)((i >> 4) & 31);
  long t = i >> 9;
  int kc = (int)(t % NCHUNK);
  int nt = (int)(t / NCHUNK);
  int lo = lane & 15, hi = lane >> 4;
  int k = kc * 32 + ((j < 8) ? (hi * 8 + j) : (16 + hi * 8 + (j - 8)));
  int n = nt * 16 + lo;
  float v = 0.0f;
  if (k < K && n < COUT) v = W[(long)k * COUT + n];
  Wp[i] = (__bf16)v;
}

__global__ void k_bnfold(const float* __restrict__ g, const float* __restrict__ be,
                         const float* __restrict__ rm, const float* __restrict__ rv,
                         float* __restrict__ sc, float* __restrict__ bi, int C) {
  int n = blockIdx.x * blockDim.x + threadIdx.x;
  if (n >= C) return;
  float s = g[n] * rsqrtf(rv[n] + EPS_BN);
  sc[n] = s;
  bi[n] = be[n] - rm[n] * s;
}

__device__ __forceinline__ void pack8(v16bf& f, int base, float4 a, float4 b) {
  f[base + 0] = (__bf16)a.x; f[base + 1] = (__bf16)a.y;
  f[base + 2] = (__bf16)a.z; f[base + 3] = (__bf16)a.w;
  f[base + 4] = (__bf16)b.x; f[base + 5] = (__bf16)b.y;
  f[base + 6] = (__bf16)b.z; f[base + 7] = (__bf16)b.w;
}

// ---------------- implicit-GEMM conv + BN + ReLU + mask + fused 2x2x2 maxpool ----------------
// Block = 128 threads = 4 waves, all sharing one 16-channel n-tile; each wave owns a
// 16-voxel m-tile. Weight chunks (1KB, fragment-ready) are double-buffered in LDS via
// gfx1250 async load-to-LDS; B-fragments come from LDS (ds_load_b128), A-fragments are
// two contiguous 32B global loads per lane (runs of 8 consecutive K live in one tap).
template <int CIN, int COUT, int S>
__global__ __launch_bounds__(128) void k_conv_wmma(
    const float* __restrict__ in, const float* __restrict__ mask3,
    const __bf16* __restrict__ Wp, const float* __restrict__ scale,
    const float* __restrict__ bias, float* __restrict__ outp) {
  constexpr int K = 27 * CIN;
  constexpr int KP = (K + 31) & ~31;
  constexpr int NCHUNK = KP / 32;
  constexpr int COUTP = (COUT + 15) & ~15;
  constexpr int Sh = S / 2;

  __shared__ __align__(32) __bf16 smem[2][512];  // 2 x 1KB weight chunks

  const int tid = threadIdx.x;
  const int wv = tid >> 5;
  const int lane = tid & 31;
  const int lo = lane & 15;
  const int hi = lane >> 4;
  const int nt = blockIdx.y;
  const long m_base = ((long)blockIdx.x * 4 + wv) * 16;  // Mtot % 64 == 0, no guards

  // row decode for this lane's A rows
  const long m = m_base + lo;
  const int x = (int)(m % S);
  const int y = (int)((m / S) % S);
  const int z = (int)((m / ((long)S * S)) % S);
  const int b = (int)(m / ((long)S * S * S));
  const int n = nt * 16 + lo;

  const char* wbytes = (const char*)(Wp + (long)nt * NCHUNK * 512);

  v8f acc = {0.f, 0.f, 0.f, 0.f, 0.f, 0.f, 0.f, 0.f};

#if ASYNC_LDS_OK
  if (tid < 64) async_copy16(wbytes + tid * 16, (char*)&smem[0][0] + tid * 16);
  wait_async();
  __syncthreads();
#endif

  for (int kc = 0; kc < NCHUNK; ++kc) {
#if ASYNC_LDS_OK
    const int buf = kc & 1;
    if (kc + 1 < NCHUNK && tid < 64)
      async_copy16(wbytes + (long)(kc + 1) * 1024 + tid * 16,
                   (char*)&smem[buf ^ 1][0] + tid * 16);
#else
    const int buf = 0;
    __syncthreads();  // protect previous iteration's readers
    if (tid < 64)
      ((float4*)&smem[0][0])[tid] = ((const float4*)(wbytes + (long)kc * 1024))[tid];
    __syncthreads();
#endif

    // ---- B fragment from LDS (ds_load_b128 x2) ----
    v16bf bfrag = *(const v16bf*)&smem[buf][lane * 16];

    // ---- A fragment ----
    v16bf afrag;
    if constexpr (CIN >= 8) {
#pragma unroll
      for (int run = 0; run < 2; ++run) {
        const int k0 = kc * 32 + run * 16 + hi * 8;  // aligned run of 8 consecutive K
        float4 va = {0.f, 0.f, 0.f, 0.f}, vb = {0.f, 0.f, 0.f, 0.f};
        if (k0 < K) {
          int tap = k0 / CIN;
          int c0 = k0 & (CIN - 1);
          int t9 = tap / 9;
          int rem = tap - t9 * 9;
          int zz = z + t9 - 1, yy = y + rem / 3 - 1, xx = x + rem - (rem / 3) * 3 - 1;
          if ((unsigned)zz < (unsigned)S && (unsigned)yy < (unsigned)S && (unsigned)xx < (unsigned)S) {
            const float4* p =
                (const float4*)&in[((((long)b * S + zz) * S + yy) * S + xx) * CIN + c0];
            va = p[0];
            vb = p[1];
          }
        }
        pack8(afrag, run * 8, va, vb);
      }
    } else {  // CIN == 1: K = 27, scalar im2col gather
#pragma unroll
      for (int j = 0; j < 16; ++j) {
        int kk = kc * 32 + ((j < 8) ? (hi * 8 + j) : (16 + hi * 8 + (j - 8)));
        float v = 0.0f;
        if (kk < K) {
          int t9 = kk / 9;
          int rem = kk - t9 * 9;
          int zz = z + t9 - 1, yy = y + rem / 3 - 1, xx = x + rem - (rem / 3) * 3 - 1;
          if ((unsigned)zz < (unsigned)S && (unsigned)yy < (unsigned)S && (unsigned)xx < (unsigned)S)
            v = in[(((long)b * S + zz) * S + yy) * S + xx];
        }
        afrag[j] = (__bf16)v;
      }
    }

    acc = __builtin_amdgcn_wmma_f32_16x16x32_bf16(false, afrag, false, bfrag,
                                                  (short)0, acc, false, false);

#if ASYNC_LDS_OK
    wait_async();
    __syncthreads();
#endif
  }

  const bool nvalid = (COUT == COUTP) || (n < COUT);
  const float sc = nvalid ? scale[n] : 0.0f;
  const float bi = nvalid ? bias[n] : 0.0f;
#pragma unroll
  for (int r = 0; r < 8; ++r) {
    if (nvalid) {
      long mm = m_base + r + 8 * hi;
      float v = fmaxf(acc[r] * sc + bi, 0.0f) * mask3[mm];  // BN -> ReLU -> mask
      int xx = (int)(mm % S);
      int yy = (int)((mm / S) % S);
      int zz = (int)((mm / ((long)S * S)) % S);
      int bb = (int)(mm / ((long)S * S * S));
      long oidx = ((((long)bb * Sh + (zz >> 1)) * Sh + (yy >> 1)) * Sh + (xx >> 1)) * COUT + n;
      // fused 2x2x2 maxpool; v >= 0 and dst zero-initialized -> int-bit max is exact
      atomicMax((unsigned int*)&outp[oidx], __float_as_uint(v));
    }
  }
}

__global__ void k_final_linear(const float* __restrict__ dense, const float* __restrict__ maskF,
                               const float* __restrict__ lw, const float* __restrict__ lb,
                               float* __restrict__ out) {
  int b = blockIdx.x, o = threadIdx.x;
  float s = 0.0f;
  for (int c = 0; c < 256; ++c) s = fmaf(dense[b * 256 + c], lw[c * 128 + o], s);
  out[b * 128 + o] = (s + lb[o]) * maskF[b];
}

// ---------------- host launcher ----------------

static inline void zero_launch(float* p, long n, hipStream_t stream) {
  long blocks = (n + 255) / 256;
  if (blocks > 8192) blocks = 8192;
  if (blocks < 1) blocks = 1;
  k_zero_f32<<<dim3((unsigned)blocks), dim3(256), 0, stream>>>(p, n);
}

extern "C" void kernel_launch(void* const* d_in, const int* in_sizes, int n_in,
                              void* d_out, int out_size, void* d_ws, size_t ws_size,
                              hipStream_t stream) {
  const int B = 8, S0 = 64;
  const int N = in_sizes[0] / 4;

  const int* coords = (const int*)d_in[0];
  const float* feat = (const float*)d_in[1];
  const float* lin_w = (const float*)d_in[32];
  const float* lin_b = (const float*)d_in[33];

  // workspace layout (floats); big buffers are exactly 8*64^3 = 2M floats each
  char* ws = (char*)d_ws;
  const long DSZ = 2L * 1024 * 1024;
  float* sums   = (float*)(ws + 0);
  float* shift  = (float*)(ws + 256);
  float* denseA = (float*)(ws + 512);
  float* denseB = denseA + DSZ;
  float* maskA  = denseB + DSZ;
  float* maskB  = maskA + DSZ;
  float* mask3  = maskB + DSZ;
  __bf16* Wp    = (__bf16*)(mask3 + DSZ);             // up to 1.77MB packed weights
  float* bnsc   = (float*)((char*)Wp + 2L * 1024 * 1024);
  float* bnbi   = bnsc + 256;

  zero_launch(sums, B * 4, stream);
  zero_launch(denseA, 8L * 64 * 64 * 64, stream);
  zero_launch(maskA, 8L * 64 * 64 * 64, stream);

  k_batch_sums<<<(N + 255) / 256, 256, 0, stream>>>(coords, sums, N);
  k_calc_shift<<<1, 32, 0, stream>>>(sums, shift, B, S0);
  k_scatter<<<(N + 255) / 256, 256, 0, stream>>>(coords, feat, shift, denseA, maskA, N, S0);

  float* din = denseA; float* dout = denseB;
  float* msk = maskA;  float* mout = maskB;

#define RUN_LAYER(I, CIN, COUT, SS)                                                                 \
  do {                                                                                              \
    const float* W     = (const float*)d_in[2 + 5 * (I) + 0];                                       \
    const float* gamma = (const float*)d_in[2 + 5 * (I) + 1];                                       \
    const float* beta  = (const float*)d_in[2 + 5 * (I) + 2];                                       \
    const float* rmean = (const float*)d_in[2 + 5 * (I) + 3];                                       \
    const float* rvar  = (const float*)d_in[2 + 5 * (I) + 4];                                       \
    const int K = 27 * (CIN);                                                                       \
    const int NCHUNK = ((K + 31) & ~31) / 32;                                                       \
    const int NT = (((COUT) + 15) & ~15) / 16;                                                      \
    long nW = (long)NT * NCHUNK * 512;                                                              \
    k_pack_w<<<(unsigned)((nW + 255) / 256), 256, 0, stream>>>(W, Wp, K, NCHUNK, (COUT), NT);       \
    k_bnfold<<<1, 256, 0, stream>>>(gamma, beta, rmean, rvar, bnsc, bnbi, (COUT));                  \
    const int Sh = (SS) / 2;                                                                        \
    long nvox  = 8L * (SS) * (SS) * (SS);                                                           \
    long nOut  = 8L * Sh * Sh * Sh * (COUT);                                                        \
    long nMOut = 8L * Sh * Sh * Sh;                                                                 \
    zero_launch(dout, nOut, stream);                                                                \
    zero_launch(mout, nMOut, stream);                                                               \
    k_mask3_pool<<<(unsigned)((nvox + 255) / 256), 256, 0, stream>>>(msk, mask3, mout, (SS), nvox); \
    dim3 grid((unsigned)(nvox / 64), NT);                                                           \
    k_conv_wmma<CIN, COUT, SS><<<grid, 128, 0, stream>>>(din, mask3, Wp, bnsc, bnbi, dout);         \
    { float* t = din; din = dout; dout = t; t = msk; msk = mout; mout = t; }                        \
  } while (0)

  RUN_LAYER(0,   1,   8, 64);
  RUN_LAYER(1,   8,  16, 32);
  RUN_LAYER(2,  16,  32, 16);
  RUN_LAYER(3,  32,  64,  8);
  RUN_LAYER(4,  64, 128,  4);
  RUN_LAYER(5, 128, 256,  2);
#undef RUN_LAYER

  // din = (8,1,1,1,256), msk = (8,1,1,1,1)
  k_final_linear<<<B, 128, 0, stream>>>(din, msk, lin_w, lin_b, (float*)d_out);

  (void)n_in; (void)out_size; (void)ws_size;
}
